// NeuronQwen35DecoderLayer_62070867361984
// MI455X (gfx1250) — matrix operand order repaired and checked by reading, not compiled, not run on hardware
//
#include <hip/hip_runtime.h>
#include <math.h>

// ---------------------------------------------------------------------------
// Qwen3.5 gated delta-rule decoder layer for MI455X (gfx1250, wave32, WMMA).
//
// Pipeline:
//   1) cvt fp32->bf16 for activations + weights
//   2) WMMA bf16 GEMMs: qkv (N=8192), z (N=4096), b||a (N=64)
//   3) causal depthwise conv (KW=4) + SiLU
//   4) l2norm(q)/sqrt(DK), l2norm(k); gates g = -exp(A_log)*softplus(a+dt_bias),
//      beta = sigmoid(b)
//   5) sequential gated delta-rule scan: 64 blocks (B*HV), 128 threads, state
//      (128x128) fully register-resident (128 VGPRs/lane), k/q via LDS bcast
//   6) per-head RMSNorm * silu(z) -> bf16
//   7) WMMA bf16 GEMM out projection -> d_out (fp32)
// ---------------------------------------------------------------------------

typedef __attribute__((ext_vector_type(16))) __bf16 v16bf;
typedef __attribute__((ext_vector_type(8)))  float  v8f;

constexpr int B_  = 2, S_ = 2048, D_ = 2048;
constexpr int HV = 32, HK = 16, DK = 128, DV = 128;
constexpr int KEY_DIM   = HK * DK;                 // 2048
constexpr int VALUE_DIM = HV * DV;                 // 4096
constexpr int CONV_DIM  = 2 * KEY_DIM + VALUE_DIM; // 8192
constexpr int KW   = 4;
constexpr int MTOT = B_ * S_;                      // 4096 rows for all GEMMs
constexpr float EPS_ = 1e-6f;

__device__ __forceinline__ unsigned short f32_to_bf16(float f) {
    unsigned u = __float_as_uint(f);
    unsigned r = u + 0x7fffu + ((u >> 16) & 1u);     // round-to-nearest-even
    if ((u & 0x7f800000u) == 0x7f800000u) r = u;     // inf/nan passthrough
    return (unsigned short)(r >> 16);
}
__device__ __forceinline__ float bf16_to_f32(unsigned short h) {
    return __uint_as_float(((unsigned)h) << 16);
}
__device__ __forceinline__ float sigmoidf_(float x) { return 1.0f / (1.0f + expf(-x)); }

// ---------------------------------------------------------------------------
// fp32 -> bf16 elementwise converter
// ---------------------------------------------------------------------------
__global__ void cvt_f32_bf16_kernel(const float* __restrict__ src,
                                    unsigned short* __restrict__ dst,
                                    long long n) {
    long long i = (long long)blockIdx.x * blockDim.x + threadIdx.x;
    if (i < n) dst[i] = f32_to_bf16(src[i]);
}

// ---------------------------------------------------------------------------
// Tiled WMMA GEMM:  C(M,N) = A(M,K) * W(N,K)^T   (bf16 in, fp32 accumulate)
//   block tile 128x128, BK=32 (exactly one v_wmma_f32_16x16x32_bf16 K-step)
//   8 waves, wave grid 2(M) x 4(N), each wave 64x32 = 4x2 accumulators.
//   LDS row stride 40 bf16 (80B): 16B-aligned rows AND conflict-free gathers.
// A-fragment (ISA 7.12.2, 16-bit A 16x32): lane L holds row m=L%16,
//   K = (L/16)*16 + {0..15} (contiguous, as B; not the ISA table order) -> two b128 LDS loads.
// B-fragment (32x16): lane L holds col n=L%16, K = (L/16)*16 + {0..15}
//   -> two contiguous b128 LDS loads.
// ---------------------------------------------------------------------------
constexpr int GBM = 128, GBN = 128, GBK = 32, LDT = 40;

union Frag16 { v16bf v; uint4 u[2]; };

template <int OUT_BF16>
__global__ __launch_bounds__(256) void gemm_bf16_kernel(
    const unsigned short* __restrict__ A,   // (M,K) bf16
    const unsigned short* __restrict__ W,   // (N,K) bf16
    void* __restrict__ Cout,                // (M,N) fp32 or bf16
    int M, int N, int K) {
    __shared__ __align__(16) unsigned short As[GBM * LDT];
    __shared__ __align__(16) unsigned short Bs[GBN * LDT];

    const int tid  = threadIdx.x;
    const int lane = tid & 31;
    const int wave = tid >> 5;     // 0..7
    const int wm   = wave & 1;     // 2 waves along M (64 rows each)
    const int wn   = wave >> 1;    // 4 waves along N (32 cols each)

    const int block_m = blockIdx.y * GBM;
    const int block_n = blockIdx.x * GBN;

    v8f acc[4][2] = {};

    const int lr = tid >> 1;            // 0..127: tile row loaded by this thread
    const int lc = (tid & 1) * 16;      // 0 or 16: K sub-column

    for (int k0 = 0; k0 < K; k0 += GBK) {
        // ---- cooperative stage to LDS (32B per thread per matrix) ----
        {
            const uint4* srcA =
                (const uint4*)(A + (size_t)(block_m + lr) * K + k0 + lc);
            uint4* dstA = (uint4*)(&As[lr * LDT + lc]);
            dstA[0] = srcA[0];
            dstA[1] = srcA[1];

            const int n = block_n + lr;
            uint4 b0 = {0, 0, 0, 0}, b1 = {0, 0, 0, 0};
            if (n < N) {
                const uint4* srcB = (const uint4*)(W + (size_t)n * K + k0 + lc);
                b0 = srcB[0];
                b1 = srcB[1];
            }
            uint4* dstB = (uint4*)(&Bs[lr * LDT + lc]);
            dstB[0] = b0;
            dstB[1] = b1;
        }
        __syncthreads();

        // ---- gather fragments ----
        Frag16 fa[4], fb[2];
        const int ko = (lane >> 4) * 16;  // A: contiguous-K base per lane half
#pragma unroll
        for (int i = 0; i < 4; ++i) {
            const unsigned short* pa =
                &As[(wm * 64 + i * 16 + (lane & 15)) * LDT];
            fa[i].u[0] = *(const uint4*)(pa + ko);
            fa[i].u[1] = *(const uint4*)(pa + 8 + ko);
        }
#pragma unroll
        for (int j = 0; j < 2; ++j) {
            const unsigned short* pb =
                &Bs[(wn * 32 + j * 16 + (lane & 15)) * LDT + (lane >> 4) * 16];
            fb[j].u[0] = ((const uint4*)pb)[0];
            fb[j].u[1] = ((const uint4*)pb)[1];
        }

        // ---- 8 WMMAs per K-step ----
#pragma unroll
        for (int i = 0; i < 4; ++i)
#pragma unroll
            for (int j = 0; j < 2; ++j)
                acc[i][j] = __builtin_amdgcn_wmma_f32_16x16x32_bf16(
                    false, fa[i].v, false, fb[j].v, (short)0, acc[i][j],
                    false, false);

        __syncthreads();
    }

    // ---- epilogue: C/D layout VGPR r, lane L -> M=r+(L/16)*8, N=L%16 ----
#pragma unroll
    for (int i = 0; i < 4; ++i) {
#pragma unroll
        for (int j = 0; j < 2; ++j) {
            const int cm = block_m + wm * 64 + i * 16 + (lane >> 4) * 8;
            const int cn = block_n + wn * 32 + j * 16 + (lane & 15);
            if (cn < N) {
#pragma unroll
                for (int r = 0; r < 8; ++r) {
                    const size_t off = (size_t)(cm + r) * N + cn;
                    if (OUT_BF16)
                        ((unsigned short*)Cout)[off] = f32_to_bf16(acc[i][j][r]);
                    else
                        ((float*)Cout)[off] = acc[i][j][r];
                }
            }
        }
    }
}

// ---------------------------------------------------------------------------
// causal depthwise conv1d (KW=4) + SiLU; qkv is bf16 (B,S,CONV_DIM)
// ---------------------------------------------------------------------------
__global__ void conv_silu_kernel(const unsigned short* __restrict__ qkv,
                                 const float* __restrict__ conv_w,
                                 float* __restrict__ conv_out) {
    const long long total = (long long)MTOT * CONV_DIM;
    long long idx = (long long)blockIdx.x * blockDim.x + threadIdx.x;
    if (idx >= total) return;
    const int c  = (int)(idx % CONV_DIM);
    const long long bs = idx / CONV_DIM;
    const int s = (int)(bs % S_);
    const long long b = bs / S_;

    float acc = 0.0f;
#pragma unroll
    for (int t = 0; t < KW; ++t) {
        const int sp = s + t - (KW - 1);
        if (sp >= 0)
            acc += bf16_to_f32(qkv[((size_t)b * S_ + sp) * CONV_DIM + c]) *
                   conv_w[c * KW + t];
    }
    conv_out[idx] = acc * sigmoidf_(acc);   // silu
}

// ---------------------------------------------------------------------------
// l2norm over DK for q (with DK^-0.5 scale) and k; one thread per (b,s,hk)
// ---------------------------------------------------------------------------
__global__ void qk_norm_kernel(const float* __restrict__ conv_out,
                               float* __restrict__ qn,
                               float* __restrict__ kn) {
    const int idx = blockIdx.x * blockDim.x + threadIdx.x;
    if (idx >= MTOT * HK) return;
    const int hk = idx & (HK - 1);
    const size_t bs = (size_t)(idx >> 4);   // HK == 16
    const float* qp = conv_out + bs * CONV_DIM + hk * DK;
    const float* kp = qp + KEY_DIM;

    float sq = 0.0f, sk = 0.0f;
#pragma unroll 8
    for (int i = 0; i < DK; ++i) { sq += qp[i] * qp[i]; sk += kp[i] * kp[i]; }
    const float qs = 0.08838834764831845f / fmaxf(sqrtf(sq), 1e-6f); // DK^-0.5
    const float ks = 1.0f / fmaxf(sqrtf(sk), 1e-6f);
#pragma unroll 8
    for (int i = 0; i < DK; ++i) {
        qn[bs * KEY_DIM + hk * DK + i] = qp[i] * qs;
        kn[bs * KEY_DIM + hk * DK + i] = kp[i] * ks;
    }
}

// ---------------------------------------------------------------------------
// gates: ba is (M,64): cols 0..31 = b, 32..63 = a
// ---------------------------------------------------------------------------
__global__ void gate_kernel(const float* __restrict__ ba,
                            const float* __restrict__ A_log,
                            const float* __restrict__ dt_bias,
                            float* __restrict__ g,
                            float* __restrict__ beta) {
    const int idx = blockIdx.x * blockDim.x + threadIdx.x;
    if (idx >= MTOT * HV) return;
    const int h = idx & (HV - 1);
    const size_t bs = (size_t)(idx >> 5);   // HV == 32
    const float bb = ba[bs * 64 + h];
    const float aa = ba[bs * 64 + 32 + h];
    const float x  = aa + dt_bias[h];
    const float sp = (x > 20.0f) ? x : log1pf(expf(x));   // softplus
    g[idx]    = -expf(A_log[h]) * sp;
    beta[idx] = sigmoidf_(bb);
}

// ---------------------------------------------------------------------------
// gated delta-rule scan: 1 block per (b,h); thread t owns DV-column t with the
// 128-entry K-slice of state in registers; k_t/q_t broadcast through LDS.
// ---------------------------------------------------------------------------
__global__ __launch_bounds__(128) void recurrence_kernel(
    const float* __restrict__ qn, const float* __restrict__ kn,
    const float* __restrict__ conv_out, const float* __restrict__ g,
    const float* __restrict__ beta, float* __restrict__ core) {
    const int bh = blockIdx.x;
    const int b  = bh >> 5;          // HV == 32
    const int h  = bh & 31;
    const int hk = h >> 1;           // rep = HV/HK = 2
    const int t  = threadIdx.x;      // DV column

    __shared__ float kls[DK];
    __shared__ float qls[DK];

    float st[DK];
#pragma unroll
    for (int i = 0; i < DK; ++i) st[i] = 0.0f;

    for (int s = 0; s < S_; ++s) {
        const size_t bs = (size_t)b * S_ + s;
        const float gg = g[bs * HV + h];
        const float bt = beta[bs * HV + h];
        const float decay = __expf(gg);

        kls[t] = kn[bs * KEY_DIM + hk * DK + t];
        qls[t] = qn[bs * KEY_DIM + hk * DK + t];
        if (s + 8 < S_) {   // gfx1250 global_prefetch_b8 hint for future rows
            __builtin_prefetch(&kn[(bs + 8) * KEY_DIM + hk * DK + t], 0, 1);
            __builtin_prefetch(
                &conv_out[(bs + 8) * CONV_DIM + 2 * KEY_DIM + h * DV + t], 0, 1);
        }
        __syncthreads();

        const float vt = conv_out[bs * CONV_DIM + 2 * KEY_DIM + h * DV + t];

        float kv = 0.0f;
#pragma unroll
        for (int i = 0; i < DK; ++i) {
            st[i] *= decay;
            kv += kls[i] * st[i];
        }
        const float delta = (vt - kv) * bt;

        float out = 0.0f;
#pragma unroll
        for (int i = 0; i < DK; ++i) {
            st[i] += kls[i] * delta;
            out += qls[i] * st[i];
        }
        core[bs * VALUE_DIM + h * DV + t] = out;
        __syncthreads();
    }
}

// ---------------------------------------------------------------------------
// per-head RMSNorm * norm_w, gated by silu(z); emits bf16 for the final GEMM
// ---------------------------------------------------------------------------
__global__ void rmsnorm_gate_kernel(const float* __restrict__ core,
                                    const float* __restrict__ z,
                                    const float* __restrict__ norm_w,
                                    unsigned short* __restrict__ core_bf) {
    const int idx = blockIdx.x * blockDim.x + threadIdx.x;
    if (idx >= MTOT * HV) return;
    const int h = idx & (HV - 1);
    const size_t bs = (size_t)(idx >> 5);
    const float* cp = core + bs * VALUE_DIM + h * DV;
    const float* zp = z + bs * VALUE_DIM + h * DV;

    float ss = 0.0f;
#pragma unroll 8
    for (int i = 0; i < DV; ++i) ss += cp[i] * cp[i];
    const float r = rsqrtf(ss * (1.0f / DV) + EPS_);
#pragma unroll 8
    for (int i = 0; i < DV; ++i) {
        const float zv = zp[i];
        const float val = cp[i] * r * norm_w[i] * (zv * sigmoidf_(zv));
        core_bf[bs * VALUE_DIM + h * DV + i] = f32_to_bf16(val);
    }
}

// ---------------------------------------------------------------------------
// launcher
// ---------------------------------------------------------------------------
extern "C" void kernel_launch(void* const* d_in, const int* in_sizes, int n_in,
                              void* d_out, int out_size, void* d_ws,
                              size_t ws_size, hipStream_t stream) {
    (void)in_sizes; (void)n_in; (void)out_size; (void)ws_size;

    const float* h       = (const float*)d_in[0];
    const float* w_qkv   = (const float*)d_in[1];
    const float* w_z     = (const float*)d_in[2];
    const float* w_b     = (const float*)d_in[3];
    const float* w_a     = (const float*)d_in[4];
    const float* conv_w  = (const float*)d_in[5];
    const float* A_log   = (const float*)d_in[6];
    const float* dt_bias = (const float*)d_in[7];
    const float* norm_w  = (const float*)d_in[8];
    const float* w_out   = (const float*)d_in[9];
    float* out = (float*)d_out;

    // workspace carve-up (~525 MB total)
    char* ws = (char*)d_ws;
    size_t off = 0;
    auto carve = [&](size_t bytes) -> void* {
        void* p = ws + off;
        off += (bytes + 255) & ~(size_t)255;
        return p;
    };
    unsigned short* h_bf    = (unsigned short*)carve((size_t)MTOT * D_ * 2);
    unsigned short* wqkv_bf = (unsigned short*)carve((size_t)CONV_DIM * D_ * 2);
    unsigned short* wz_bf   = (unsigned short*)carve((size_t)VALUE_DIM * D_ * 2);
    unsigned short* wba_bf  = (unsigned short*)carve((size_t)64 * D_ * 2);
    unsigned short* wout_bf = (unsigned short*)carve((size_t)D_ * VALUE_DIM * 2);
    unsigned short* qkv_bf  = (unsigned short*)carve((size_t)MTOT * CONV_DIM * 2);
    float* z_f      = (float*)carve((size_t)MTOT * VALUE_DIM * 4);
    float* ba_f     = (float*)carve((size_t)MTOT * 64 * 4);
    float* conv_out = (float*)carve((size_t)MTOT * CONV_DIM * 4);
    float* qn       = (float*)carve((size_t)MTOT * KEY_DIM * 4);
    float* kn       = (float*)carve((size_t)MTOT * KEY_DIM * 4);
    float* g_buf    = (float*)carve((size_t)MTOT * HV * 4);
    float* beta_buf = (float*)carve((size_t)MTOT * HV * 4);
    float* core     = (float*)carve((size_t)MTOT * VALUE_DIM * 4);
    unsigned short* core_bf = (unsigned short*)carve((size_t)MTOT * VALUE_DIM * 2);

    auto cvt = [&](const float* s, unsigned short* d, long long n) {
        cvt_f32_bf16_kernel<<<(unsigned)((n + 255) / 256), 256, 0, stream>>>(s, d, n);
    };
    cvt(h,     h_bf,    (long long)MTOT * D_);
    cvt(w_qkv, wqkv_bf, (long long)CONV_DIM * D_);
    cvt(w_z,   wz_bf,   (long long)VALUE_DIM * D_);
    cvt(w_b,   wba_bf,             (long long)HV * D_);   // rows 0..31
    cvt(w_a,   wba_bf + (size_t)HV * D_, (long long)HV * D_); // rows 32..63
    cvt(w_out, wout_bf, (long long)D_ * VALUE_DIM);

    const dim3 gthr(256);
    // qkv: (4096 x 8192) = (4096 x 2048)*(8192 x 2048)^T, bf16 out for conv
    gemm_bf16_kernel<1><<<dim3(CONV_DIM / GBN, MTOT / GBM), gthr, 0, stream>>>(
        h_bf, wqkv_bf, qkv_bf, MTOT, CONV_DIM, D_);
    // z: (4096 x 4096)
    gemm_bf16_kernel<0><<<dim3(VALUE_DIM / GBN, MTOT / GBM), gthr, 0, stream>>>(
        h_bf, wz_bf, z_f, MTOT, VALUE_DIM, D_);
    // b||a: (4096 x 64), N guarded inside the kernel
    gemm_bf16_kernel<0><<<dim3(1, MTOT / GBM), gthr, 0, stream>>>(
        h_bf, wba_bf, ba_f, MTOT, 64, D_);

    {
        const long long n = (long long)MTOT * CONV_DIM;
        conv_silu_kernel<<<(unsigned)((n + 255) / 256), 256, 0, stream>>>(
            qkv_bf, conv_w, conv_out);
    }
    qk_norm_kernel<<<(MTOT * HK + 255) / 256, 256, 0, stream>>>(conv_out, qn, kn);
    gate_kernel<<<(MTOT * HV + 255) / 256, 256, 0, stream>>>(
        ba_f, A_log, dt_bias, g_buf, beta_buf);

    recurrence_kernel<<<B_ * HV, DV, 0, stream>>>(qn, kn, conv_out, g_buf,
                                                  beta_buf, core);

    rmsnorm_gate_kernel<<<(MTOT * HV + 255) / 256, 256, 0, stream>>>(
        core, z_f, norm_w, core_bf);

    // out: (4096 x 2048) = core_bf (4096 x 4096) * w_out (2048 x 4096)^T
    gemm_bf16_kernel<0><<<dim3(D_ / GBN, MTOT / GBM), gthr, 0, stream>>>(
        core_bf, wout_bf, out, MTOT, D_, VALUE_DIM);
}
